// VectorQuantizer_90675349553702
// MI455X (gfx1250) — matrix operand order repaired and checked by reading, not compiled, not run on hardware
//
#include <hip/hip_runtime.h>
#include <hip/hip_bf16.h>

// ---------------------------------------------------------------------------
// VQ-VAE vector quantizer for MI455X (gfx1250, wave32, WMMA).
//
// N = 131072 rows (d=64), C = 1024 codes.
// dist = ||z||^2 - 2 z.c + ||c||^2 ; argmin & softmax(-dist) are invariant to
// the per-row ||z||^2 term and to a positive rescale of the logits when the
// exponential base is changed consistently, so we compute
//   G = log2(e) * (2 z.c - ||c||^2)
// with v_wmma_f32_16x16x32_bf16 (hi/lo bf16 split of both operands: 3 WMMA
// products -> ~fp32 accuracy) and use exp2 (bare v_exp_f32) everywhere.
// Losses/quantized output are computed exactly from the f32 inputs.
// ---------------------------------------------------------------------------

typedef __attribute__((ext_vector_type(16))) __bf16 v16bf;
typedef __attribute__((ext_vector_type(8)))  float  v8f;

#define DIMK 64
#define LOG2E 1.4426950408889634f

// ---------------- Kernel 0: codebook prep + accumulator zeroing -------------
__global__ void vq_prep(const float* __restrict__ cb,
                        __bf16* __restrict__ cb_hi, __bf16* __restrict__ cb_lo,
                        float* __restrict__ cns_neg,
                        float* __restrict__ counts, float* __restrict__ soft_acc,
                        float* __restrict__ mse_acc, int C) {
  int i = blockIdx.x * blockDim.x + threadIdx.x;
  if (i >= C) return;
  const float* row = cb + (size_t)i * DIMK;
  float cn = 0.f;
#pragma unroll
  for (int k = 0; k < DIMK; ++k) {
    float v = row[k];
    cn += v * v;
    // fold the (-2 z.c) factor *and* the log2(e) logit scale into B
    float tv = 2.0f * LOG2E * v;
    __bf16 h = (__bf16)tv;
    cb_hi[(size_t)i * DIMK + k] = h;
    cb_lo[(size_t)i * DIMK + k] = (__bf16)(tv - (float)h);
  }
  cns_neg[i]  = -cn * LOG2E;           // so G = acc + cns_neg (one v_add)
  counts[i]   = 0.f;
  soft_acc[i] = 0.f;
  if (i == 0) *mse_acc = 0.f;
}

// ---------------- Kernel 1: main distance/argmin/softmax kernel -------------
__global__ __launch_bounds__(256)
void vq_main(const float* __restrict__ z, const float* __restrict__ cbf32,
             const __bf16* __restrict__ cb_hi, const __bf16* __restrict__ cb_lo,
             const float* __restrict__ cns_neg,
             float* __restrict__ out_q, float* __restrict__ out_idx,
             float* __restrict__ counts_g, float* __restrict__ soft_g,
             float* __restrict__ mse_g, int C) {
  __shared__ float s_soft[1024];
  __shared__ int   s_idx[8][16];

  const int tid   = threadIdx.x;
  const int lane  = tid & 31;
  const int wave  = tid >> 5;
  const int n16   = lane & 15;      // column (code within tile) / A row
  const int hi16  = lane >> 4;      // lane group 0/1
  const int koff  = hi16 * 16;      // A-matrix per-group K sub-offset
  const long waveRow0 = (long)blockIdx.x * 128 + (long)wave * 16;

  for (int j = tid; j < C; j += 256) s_soft[j] = 0.f;

  // ---- A fragments: 16 rows x 64 K, split into bf16 hi/lo, two K-chunks ----
  // A 16x32 as loaded here (contiguous k = 16h + s): lanes 0-15 hold M=lane, K = 0..15;
  // lanes 16-31 hold M=lane-16, K = 16..31.
  const float* zr = z + (size_t)(waveRow0 + n16) * DIMK;
  v16bf a_hi[2], a_lo[2];
#pragma unroll
  for (int c2 = 0; c2 < 2; ++c2) {
    const int kb = c2 * 32;
#pragma unroll
    for (int i = 0; i < 8; ++i) {
      float f0 = zr[kb + koff + i];
      float f1 = zr[kb + koff + 8 + i];
      __bf16 h0 = (__bf16)f0, h1 = (__bf16)f1;
      a_hi[c2][i]     = h0;  a_lo[c2][i]     = (__bf16)(f0 - (float)h0);
      a_hi[c2][8 + i] = h1;  a_lo[c2][8 + i] = (__bf16)(f1 - (float)h1);
    }
  }

  // ---- Pass 1: branchless online (max, sum-exp2, argmax) over code tiles ---
  float m[8], s[8]; int bi[8];
#pragma unroll
  for (int v = 0; v < 8; ++v) { m[v] = -3.0e38f; s[v] = 0.f; bi[v] = 0; }

  const int ntiles = C >> 4;
  for (int t = 0; t < ntiles; ++t) {
    const int col = (t << 4) + n16;
    // B 32x16, same contiguous k order as A: lanes 0-15 hold N=lane, K=0..15; lanes 16-31 K=16..31.
    const __bf16* bh = cb_hi + (size_t)col * DIMK + hi16 * 16;
    const __bf16* bl = cb_lo + (size_t)col * DIMK + hi16 * 16;
    v8f acc = {};
#pragma unroll
    for (int c2 = 0; c2 < 2; ++c2) {
      v16bf bhv = *(const v16bf*)(bh + c2 * 32);
      v16bf blv = *(const v16bf*)(bl + c2 * 32);
      acc = __builtin_amdgcn_wmma_f32_16x16x32_bf16(false, a_hi[c2], false, bhv,
                                                    (short)0, acc, false, false);
      acc = __builtin_amdgcn_wmma_f32_16x16x32_bf16(false, a_hi[c2], false, blv,
                                                    (short)0, acc, false, false);
      acc = __builtin_amdgcn_wmma_f32_16x16x32_bf16(false, a_lo[c2], false, bhv,
                                                    (short)0, acc, false, false);
    }
    const float cns = cns_neg[col];
#pragma unroll
    for (int v = 0; v < 8; ++v) {           // acc[v] = logit2[row v+8*hi16][n16]
      float G  = acc[v] + cns;
      float nm = fmaxf(m[v], G);
      // branchless online update; exp2 -> bare v_exp_f32 (TRANS, co-exec w/ WMMA)
      s[v]  = s[v] * exp2f(m[v] - nm) + exp2f(G - nm);
      bi[v] = (G > m[v]) ? col : bi[v];     // strict > keeps first (lowest) index
      m[v]  = nm;
    }
  }

  // ---- Cross-lane reduce over the 16 lanes holding each row ----
#pragma unroll
  for (int v = 0; v < 8; ++v) {
    float mv = m[v], sv = s[v]; int iv = bi[v];
#pragma unroll
    for (int off = 1; off < 16; off <<= 1) {
      float om = __shfl_xor(mv, off, 32);
      float os = __shfl_xor(sv, off, 32);
      int   oi = __shfl_xor(iv, off, 32);
      float nm = fmaxf(mv, om);
      sv = sv * exp2f(mv - nm) + os * exp2f(om - nm);
      iv = (om > mv || (om == mv && oi < iv)) ? oi : iv;  // jnp argmin tie: low idx
      mv = nm;
    }
    m[v] = mv; s[v] = 1.0f / sv; bi[v] = iv;
    if (n16 == v) {                         // one writer per row
      int  rw   = v + 8 * hi16;
      long grow = waveRow0 + rw;
      s_idx[wave][rw] = iv;
      out_idx[grow]   = (float)iv;
      atomicAdd(&counts_g[iv], 1.0f);
    }
  }

  __syncthreads();   // s_soft zeroed before pass-2 accumulation

  // ---- Pass 2: recompute tiles, accumulate softmax probs per code ----
  for (int t = 0; t < ntiles; ++t) {
    const int col = (t << 4) + n16;
    const __bf16* bh = cb_hi + (size_t)col * DIMK + hi16 * 16;
    const __bf16* bl = cb_lo + (size_t)col * DIMK + hi16 * 16;
    v8f acc = {};
#pragma unroll
    for (int c2 = 0; c2 < 2; ++c2) {
      v16bf bhv = *(const v16bf*)(bh + c2 * 32);
      v16bf blv = *(const v16bf*)(bl + c2 * 32);
      acc = __builtin_amdgcn_wmma_f32_16x16x32_bf16(false, a_hi[c2], false, bhv,
                                                    (short)0, acc, false, false);
      acc = __builtin_amdgcn_wmma_f32_16x16x32_bf16(false, a_hi[c2], false, blv,
                                                    (short)0, acc, false, false);
      acc = __builtin_amdgcn_wmma_f32_16x16x32_bf16(false, a_lo[c2], false, bhv,
                                                    (short)0, acc, false, false);
    }
    const float cns = cns_neg[col];
    float local = 0.f;
#pragma unroll
    for (int v = 0; v < 8; ++v)
      local += exp2f((acc[v] + cns) - m[v]) * s[v];  // prob contribution, 8 rows
    atomicAdd(&s_soft[col], local);                  // ds_add_f32
  }

  // ---- Quantized output + exact MSE from f32 data ----
  float sumsq = 0.f;
  for (int r = 0; r < 16; ++r) {
    const int   gi   = s_idx[wave][r];
    const long  grow = waveRow0 + r;
    const float* crow = cbf32 + (size_t)gi * DIMK;
    const float* zrow = z + (size_t)grow * DIMK;
    float* orow = out_q + (size_t)grow * DIMK;
#pragma unroll
    for (int h = 0; h < 2; ++h) {
      int k = h * 32 + lane;
      float cv = crow[k];
      float d  = cv - zrow[k];
      orow[k] = cv;                         // quantized_st == quantized numerically
      sumsq += d * d;
    }
  }
#pragma unroll
  for (int off = 1; off < 32; off <<= 1) sumsq += __shfl_xor(sumsq, off, 32);
  if (lane == 0) atomicAdd(mse_g, sumsq);

  __syncthreads();
  for (int j = tid; j < C; j += 256) atomicAdd(&soft_g[j], s_soft[j]);
}

// ---------------- Kernel 2: finalize scalars / usage / perplexity -----------
__global__ __launch_bounds__(1024)
void vq_finalize(const float* __restrict__ counts, const float* __restrict__ soft_acc,
                 const float* __restrict__ mse_acc,
                 float* out_loss, float* out_cb, float* out_commit, float* out_ppl,
                 float* __restrict__ out_usage, float* __restrict__ out_soft,
                 int C, long N) {
  __shared__ float red[1024];
  const int t = threadIdx.x;
  float c = (t < C) ? counts[t] : 0.f;
  red[t] = c;
  __syncthreads();
  for (int off = (int)blockDim.x >> 1; off > 0; off >>= 1) {
    if (t < off) red[t] += red[t + off];
    __syncthreads();
  }
  float total = red[0];
  __syncthreads();
  float usage = c / fmaxf(total, 1.0f);
  if (t < C) out_usage[t] = usage;
  red[t] = usage * logf(usage + 1e-8f);
  __syncthreads();
  for (int off = (int)blockDim.x >> 1; off > 0; off >>= 1) {
    if (t < off) red[t] += red[t + off];
    __syncthreads();
  }
  if (t < C) out_soft[t] = soft_acc[t] / (float)N;
  if (t == 0) {
    float mse = *mse_acc / (float)(N * 64);
    out_cb[0]     = mse;
    out_commit[0] = mse;                    // numerically identical
    out_loss[0]   = mse * 1.25f;            // mse + 0.25*mse
    out_ppl[0]    = __expf(-red[0]);
  }
}

// ---------------------------------------------------------------------------
extern "C" void kernel_launch(void* const* d_in, const int* in_sizes, int n_in,
                              void* d_out, int out_size, void* d_ws, size_t ws_size,
                              hipStream_t stream) {
  const float* z  = (const float*)d_in[0];
  const float* cb = (const float*)d_in[1];
  const long N = (long)in_sizes[0] / DIMK;   // 131072
  const int  C = in_sizes[1] / DIMK;         // 1024

  // workspace layout (~270 KB)
  char* w = (char*)d_ws;
  __bf16* cb_hi = (__bf16*)w;                               // C*64 bf16
  __bf16* cb_lo = (__bf16*)(w + (size_t)C * DIMK * 2);      // C*64 bf16
  float*  cnsng = (float*)(w + (size_t)C * DIMK * 4);
  float*  counts = cnsng + C;
  float*  softa  = counts + C;
  float*  msea   = softa + C;

  // output layout (flattened tuple, all float)
  float* out        = (float*)d_out;
  float* out_q      = out;                         // N*64
  float* out_loss   = out + (size_t)N * DIMK;      // 1
  float* out_idx    = out_loss + 1;                // N
  float* out_cb     = out_idx + N;                 // 1
  float* out_commit = out_cb + 1;                  // 1
  float* out_ppl    = out_commit + 1;              // 1
  float* out_usage  = out_ppl + 1;                 // C
  float* out_soft   = out_usage + C;               // C

  vq_prep<<<(C + 255) / 256, 256, 0, stream>>>(cb, cb_hi, cb_lo, cnsng,
                                               counts, softa, msea, C);
  vq_main<<<(int)(N / 128), 256, 0, stream>>>(z, cb, cb_hi, cb_lo, cnsng,
                                              out_q, out_idx, counts, softa, msea, C);
  vq_finalize<<<1, 1024, 0, stream>>>(counts, softa, msea, out_loss, out_cb,
                                      out_commit, out_ppl, out_usage, out_soft, C, N);
}
